// HGDNN_75084618269161
// MI455X (gfx1250) — matrix operand order repaired and checked by reading, not compiled, not run on hardware
//
#include <hip/hip_runtime.h>
#include <hip/hip_bf16.h>
#include <math.h>

typedef __bf16 bf16;
typedef __attribute__((ext_vector_type(16))) __bf16 v16bf;
typedef __attribute__((ext_vector_type(8)))  __bf16 v8bf;
typedef __attribute__((ext_vector_type(8)))  float  v8f;

#define NN   4096
#define EE   5
#define WIN  512
#define WOUT 128
#define NCLS 16
#define MM   1024
#define NLAYERS 64
#define H0   1152   // WIN + EE*WOUT

struct BfPair { v8bf lo, hi; };

__device__ __forceinline__ v16bf ld_frag(const bf16* p, int second_off) {
  BfPair r;
  r.lo = *(const v8bf*)p;
  r.hi = *(const v8bf*)(p + second_off);
  return __builtin_bit_cast(v16bf, r);
}

__device__ __forceinline__ v8f wmma_bf16f32(v16bf a, v16bf b, v8f c) {
  // 8 args: (neg_a, A, neg_b, B, c_mod, C, reuse_a, reuse_b)
  return __builtin_amdgcn_wmma_f32_16x16x32_bf16(false, a, false, b, (short)0, c,
                                                 false, false);
}

// ---------------------------------------------------------------------------
// Generic fused bf16 GEMM: out = epilogue( Am[M,K] @ Bt[N,K]^T ),  K % 64 == 0.
//   Am row-major bf16 (A-operand), Bt is B stored transposed [N,K] bf16.
//   epilogue: v = s_acc*acc + bias[n] + (rowAdd? rowAdd[m] : s_aux)*aux[m,n];
//             if(rowScale) v*=rowScale[m]; act: 0=none 1=relu 2=leaky(0.01)
//   Writes any of: outF (f32 [M,N]), outB (bf16 [M,N]), outBT (bf16 [N,M]).
// Each wave computes a 16 x (16*NT) strip (NT WMMA tiles, A-frag reuse x NT).
// NT is a template param (no EXEC masking around WMMA); wave id is
// readfirstlane'd so tile coords live in SGPRs. The k-loop is a 2-stage
// ping-pong with the LAST iteration PEELED: the steady-state body is a single
// basic block (compute even / prefetch even@k / compute odd / prefetch
// odd@k+32) so accumulators keep fixed registers (no PHI copies, no
// WMMA->VALU-mov hazard NOPs) and loads stay a full stage ahead of use.
// ---------------------------------------------------------------------------
template <int NT>
__global__ __launch_bounds__(128) void gemm_bf16_k(
    const bf16* __restrict__ Am, int lda,
    const bf16* __restrict__ Bt, int ldb,
    int M, int N, int K,
    float s_acc, float s_aux, int act,
    const float* __restrict__ bias,
    const float* __restrict__ rowScale,
    const float* __restrict__ rowAdd,
    const float* __restrict__ aux,
    float* __restrict__ outF,
    bf16* __restrict__ outB,
    bf16* __restrict__ outBT)
{
  const int lane = threadIdx.x & 31;
  const int wave = __builtin_amdgcn_readfirstlane(threadIdx.x) >> 5;  // scalar
  const int Nt   = N / (16 * NT);
  const int tiles = (M >> 4) * Nt;
  const int tid  = blockIdx.x * 4 + wave;                             // scalar
  if (tid >= tiles) return;                                           // uniform
  const int m0 = (tid / Nt) << 4;
  const int n0 = (tid % Nt) * (16 * NT);
  const int hi = (lane >> 4) & 1;
  const int lr = lane & 15;

  // A fragment (16x32 bf16): lane lr holds row m0+lr; lanes<16: K 0..15,
  // lanes>=16: K 16..31  -> two 16B chunks at +0,+8 from (k + hi*16).
  const bf16* ap = Am + (size_t)(m0 + lr) * lda + (hi ? 16 : 0);
  // B fragment (32x16): lane lr = column n; lanes<16: K 0..15, lanes>=16: K 16..31
  // -> 16 contiguous bf16 from Bt row n at (k + hi*16): two 16B chunks at +0,+8.
  const bf16* bp[NT];
  #pragma unroll
  for (int j = 0; j < NT; ++j)
    bp[j] = Bt + (size_t)(n0 + 16 * j + lr) * ldb + (hi ? 16 : 0);

  v8f acc[NT];
  #pragma unroll
  for (int j = 0; j < NT; ++j) acc[j] = (v8f){};

  // 2-stage ping-pong prologue: even group at k=0, odd group at k=32
  v16bf a_e = ld_frag(ap, 8);
  v16bf b_e[NT];
  #pragma unroll
  for (int j = 0; j < NT; ++j) b_e[j] = ld_frag(bp[j], 8);
  v16bf a_o = ld_frag(ap + 32, 8);
  v16bf b_o[NT];
  #pragma unroll
  for (int j = 0; j < NT; ++j) b_o[j] = ld_frag(bp[j] + 32, 8);

  // steady state: single basic block, unconditional prefetch (last iter peeled)
  for (int k = 64; k < K; k += 64) {       // K % 64 == 0
    #pragma unroll
    for (int j = 0; j < NT; ++j) acc[j] = wmma_bf16f32(a_e, b_e[j], acc[j]);
    a_e = ld_frag(ap + k, 8);
    #pragma unroll
    for (int j = 0; j < NT; ++j) b_e[j] = ld_frag(bp[j] + k, 8);
    #pragma unroll
    for (int j = 0; j < NT; ++j) acc[j] = wmma_bf16f32(a_o, b_o[j], acc[j]);
    a_o = ld_frag(ap + k + 32, 8);
    #pragma unroll
    for (int j = 0; j < NT; ++j) b_o[j] = ld_frag(bp[j] + k + 32, 8);
  }
  // peeled final 64 K-elements: compute only, no loads
  #pragma unroll
  for (int j = 0; j < NT; ++j) acc[j] = wmma_bf16f32(a_e, b_e[j], acc[j]);
  #pragma unroll
  for (int j = 0; j < NT; ++j) acc[j] = wmma_bf16f32(a_o, b_o[j], acc[j]);

  #pragma unroll
  for (int j = 0; j < NT; ++j) {
    int n = n0 + 16 * j + lr;
    #pragma unroll
    for (int v = 0; v < 8; ++v) {
      int m = m0 + v + hi * 8;   // C/D layout: VGPR v = row v (+8 for hi half)
      float val = s_acc * acc[j][v];
      if (bias)     val += bias[n];
      if (aux)      val += (rowAdd ? rowAdd[m] : s_aux) * aux[(size_t)m * N + n];
      if (rowScale) val *= rowScale[m];
      if (act == 1)      val = fmaxf(val, 0.f);
      else if (act == 2) val = (val >= 0.f) ? val : 0.01f * val;
      if (outF)  outF[(size_t)m * N + n]  = val;
      if (outB)  outB[(size_t)m * N + n]  = (bf16)val;
      if (outBT) outBT[(size_t)n * M + m] = (bf16)val;
    }
  }
}

// --- weight transpose+convert: in f32 [B,R,C] -> out bf16 [B,C,R] -------------
__global__ void twK(const float* __restrict__ in, bf16* __restrict__ out,
                    int B, int R, int C) {
  size_t idx = (size_t)blockIdx.x * 256 + threadIdx.x;
  size_t tot = (size_t)B * R * C;
  if (idx >= tot) return;
  int c = (int)(idx % C);
  size_t q = idx / C;
  int r = (int)(q % R);
  int b = (int)(q / R);
  out[((size_t)b * C + c) * R + r] = (bf16)in[idx];
}

// --- elementwise f32 -> bf16 --------------------------------------------------
__global__ void convK(const float* __restrict__ in, bf16* __restrict__ o, int n) {
  int i = blockIdx.x * 256 + threadIdx.x;
  if (i < n) o[i] = (bf16)in[i];
}

// --- column sums of A over first index: colsum[n*5+e] = sum_i A[i,n,e] --------
__global__ void colsumK(const float* __restrict__ A, float* __restrict__ colsum) {
  int c = blockIdx.x * 256 + threadIdx.x;   // c = n*EE + e, coalesced
  const float* p = A + c;
  float s = 0.f;
  #pragma unroll 8
  for (int i = 0; i < NN; ++i) s += p[(size_t)i * (NN * EE)];
  colsum[c] = s;
}

// --- dinv[e][n] = 1/deg, rowadd[e][n] = 1 - A[n,n,e] --------------------------
__global__ void dinvK(const float* __restrict__ A, const float* __restrict__ colsum,
                      float* __restrict__ dinv, float* __restrict__ rowadd) {
  int idx = blockIdx.x * 256 + threadIdx.x;   // n*5+e
  int n = idx / EE, e = idx % EE;
  float diag = A[((size_t)n * NN + n) * EE + e];
  float deg  = 1.f + colsum[idx] - diag;      // deg of Ht row n
  dinv[e * NN + n]   = (deg > 0.f) ? 1.f / deg : 0.f;
  rowadd[e * NN + n] = 1.f - diag;
}

// --- transpose A -> 5 bf16 planes At_e[n][m]=A[m,n,e]; emit B-pattern bitmask -
__global__ void transposeAK(const float* __restrict__ A, bf16* __restrict__ At,
                            unsigned* __restrict__ Bmask) {
  __shared__ bf16 tile[EE][32][33];
  int m0 = blockIdx.x << 5, n0 = blockIdx.y << 5;
  int t = threadIdx.x;
  #pragma unroll
  for (int ph = 0; ph < 4; ++ph) {
    int idx = ph * 256 + t;
    int ml = idx >> 5, nl = idx & 31;         // lanes sweep nl (coalesced 640B)
    const float* src = A + ((size_t)(m0 + ml) * NN + (n0 + nl)) * EE;
    float a0 = src[0], a1 = src[1], a2 = src[2], a3 = src[3], a4 = src[4];
    tile[0][nl][ml] = (bf16)a0; tile[1][nl][ml] = (bf16)a1;
    tile[2][nl][ml] = (bf16)a2; tile[3][nl][ml] = (bf16)a3;
    tile[4][nl][ml] = (bf16)a4;
    bool nz = (a0 != 0.f) || (a1 != 0.f) || (a2 != 0.f) || (a3 != 0.f) || (a4 != 0.f);
    unsigned w = __builtin_amdgcn_ballot_w32(nz);   // 32 lanes = one mask word
    if ((t & 31) == 0) Bmask[(size_t)(m0 + ml) * 128 + (n0 >> 5)] = w;
  }
  __syncthreads();
  #pragma unroll
  for (int e = 0; e < EE; ++e) {
    bf16* dst = At + (size_t)e * NN * NN;
    #pragma unroll
    for (int ph = 0; ph < 4; ++ph) {
      int idx = ph * 256 + t;
      int nl = idx >> 5, ml = idx & 31;       // lanes sweep ml: coalesced stores
      dst[(size_t)(n0 + nl) * NN + m0 + ml] = tile[e][nl][ml];
    }
  }
}

// --- dis[n] = rsqrt(degree of Bs row n) --------------------------------------
__global__ void degreeK(const unsigned* __restrict__ Bmask, float* __restrict__ dis) {
  int n = blockIdx.x, t = threadIdx.x;        // 128 threads = 128 mask words
  unsigned w = Bmask[(size_t)n * 128 + t];
  if (t == (n >> 5)) w |= 1u << (n & 31);     // self loop
  __shared__ int red[128];
  red[t] = __popc(w);
  __syncthreads();
  for (int s = 64; s > 0; s >>= 1) { if (t < s) red[t] += red[t + s]; __syncthreads(); }
  if (t == 0) dis[n] = rsqrtf((float)red[0]);
}

// --- A_hat[n][m] = dis[n]*dis[m]*Bs[n,m] (bf16) ------------------------------
__global__ void ahatK(const unsigned* __restrict__ Bmask, const float* __restrict__ dis,
                      bf16* __restrict__ Ahat) {
  int idx = blockIdx.x * 256 + threadIdx.x;   // n*128 + word
  int n = idx >> 7, j = idx & 127;
  unsigned w = Bmask[idx];
  if (j == (n >> 5)) w |= 1u << (n & 31);
  float dn = dis[n];
  __attribute__((aligned(16))) bf16 tmp[32];
  #pragma unroll
  for (int b = 0; b < 32; ++b)
    tmp[b] = ((w >> b) & 1u) ? (bf16)(dn * dis[j * 32 + b]) : (bf16)0.f;
  v8bf* dst = (v8bf*)(Ahat + (size_t)n * NN + j * 32);
  const v8bf* s = (const v8bf*)tmp;
  dst[0] = s[0]; dst[1] = s[1]; dst[2] = s[2]; dst[3] = s[3];
}

// --- attention: partial sums of attw[n]*lgcn[e,n,d] over 128-row chunks -------
__global__ void tmpartK(const float* __restrict__ lgcn, const float* __restrict__ attw,
                        float* __restrict__ tmpart) {
  int e = blockIdx.x, ch = blockIdx.y, d = threadIdx.x;
  const float* L = lgcn + ((size_t)e * NN + ch * 128) * WOUT + d;
  const float* w = attw + ch * 128;
  float s = 0.f;
  #pragma unroll 4
  for (int n = 0; n < 128; ++n) s += w[n] * L[(size_t)n * WOUT];
  tmpart[((size_t)e * 32 + ch) * WOUT + d] = s;
}

__global__ void tmfinK(const float* __restrict__ tmpart, const float* __restrict__ attb,
                       float* __restrict__ tm) {
  int e = blockIdx.x, d = threadIdx.x;
  float s = 0.f;
  for (int c = 0; c < 32; ++c) s += tmpart[((size_t)e * 32 + c) * WOUT + d];
  tm[e * WOUT + d] = tanhf(s + attb[d]);
}

// --- watt[e] = tm[e].attq ; beta = softmax(watt)*E ---------------------------
__global__ void betaK(const float* __restrict__ tm, const float* __restrict__ attq,
                      float* __restrict__ beta) {
  __shared__ float red[128];
  __shared__ float w[EE];
  int t = threadIdx.x;
  for (int e = 0; e < EE; ++e) {
    red[t] = tm[e * WOUT + t] * attq[t];
    __syncthreads();
    for (int s = 64; s > 0; s >>= 1) { if (t < s) red[t] += red[t + s]; __syncthreads(); }
    if (t == 0) w[e] = red[0];
    __syncthreads();
  }
  if (t == 0) {
    float mx = w[0];
    for (int e = 1; e < EE; ++e) mx = fmaxf(mx, w[e]);
    float s = 0.f, ex[EE];
    for (int e = 0; e < EE; ++e) { ex[e] = expf(w[e] - mx); s += ex[e]; }
    for (int e = 0; e < EE; ++e) beta[e] = (float)EE * ex[e] / s;
  }
}

// --- X_ = [beta_e * lgcn | X] as bf16 [N, 1152] ------------------------------
__global__ void concatK(const float* __restrict__ lgcn, const float* __restrict__ X,
                        const float* __restrict__ beta, bf16* __restrict__ Xcb) {
  size_t idx = (size_t)blockIdx.x * 256 + threadIdx.x;
  int n = (int)(idx / H0), h = (int)(idx % H0);
  float v;
  if (h < EE * WOUT) {
    int e = h >> 7, d = h & 127;
    v = beta[e] * lgcn[((size_t)e * NN + n) * WOUT + d];
  } else {
    v = X[(size_t)n * WIN + (h - EE * WOUT)];
  }
  Xcb[idx] = (bf16)v;
}

// --- gather labeled nodes -----------------------------------------------------
__global__ void gatherK(const bf16* __restrict__ Z, const int* __restrict__ tx,
                        bf16* __restrict__ Zg) {
  int idx = blockIdx.x * 256 + threadIdx.x;
  int i = idx >> 7, d = idx & 127;
  Zg[idx] = Z[(size_t)tx[i] * WOUT + d];
}

// --- per-row log-prob of target class ----------------------------------------
__global__ void lossrowK(const float* __restrict__ y, const int* __restrict__ tgt,
                         float* __restrict__ lrow) {
  int r = blockIdx.x * 256 + threadIdx.x;
  if (r >= MM) return;
  const float* yr = y + (size_t)r * NCLS;
  float mx = yr[0];
  for (int c = 1; c < NCLS; ++c) mx = fmaxf(mx, yr[c]);
  float s = 0.f;
  for (int c = 0; c < NCLS; ++c) s += expf(yr[c] - mx);
  lrow[r] = yr[tgt[r]] - mx - logf(s);
}

__global__ void loss2K(const float* __restrict__ lrow, float* __restrict__ out) {
  __shared__ float red[256];
  int t = threadIdx.x;
  red[t] = lrow[t] + lrow[t + 256] + lrow[t + 512] + lrow[t + 768];
  __syncthreads();
  for (int s = 128; s > 0; s >>= 1) { if (t < s) red[t] += red[t + s]; __syncthreads(); }
  if (t == 0) out[0] = -red[0] / (float)MM;
}

// ===========================================================================
extern "C" void kernel_launch(void* const* d_in, const int* in_sizes, int n_in,
                              void* d_out, int out_size, void* d_ws, size_t ws_size,
                              hipStream_t stream) {
  (void)in_sizes; (void)n_in; (void)out_size; (void)ws_size;
  const float* A      = (const float*)d_in[0];
  const float* X      = (const float*)d_in[1];
  const int*   tgtx   = (const int*)d_in[2];
  const int*   tgt    = (const int*)d_in[3];
  const float* weight = (const float*)d_in[4];
  const float* attw   = (const float*)d_in[5];
  const float* attb   = (const float*)d_in[6];
  const float* attq   = (const float*)d_in[7];
  const float* Wg     = (const float*)d_in[8];
  const float* bg     = (const float*)d_in[9];
  const float* W0     = (const float*)d_in[10];
  const float* b0     = (const float*)d_in[11];
  const float* Wconvs = (const float*)d_in[12];
  const float* Wd1    = (const float*)d_in[13];
  const float* bd1    = (const float*)d_in[14];
  const float* W1     = (const float*)d_in[15];
  const float* b1     = (const float*)d_in[16];
  const float* W2     = (const float*)d_in[17];
  const float* b2     = (const float*)d_in[18];
  float* out = (float*)d_out;

  char* p = (char*)d_ws;
  auto bump = [&](size_t bytes) -> char* {
    char* r = p; p += (bytes + 255) & ~(size_t)255; return r;
  };
  bf16*     At      = (bf16*)bump((size_t)EE * NN * NN * 2);        // 167.8 MB
  bf16*     Ahat    = (bf16*)bump((size_t)NN * NN * 2);             // 33.6 MB (L2-resident)
  unsigned* Bmask   = (unsigned*)bump((size_t)NN * 128 * 4);
  float*    colsum  = (float*)bump((size_t)NN * EE * 4);
  float*    dinv    = (float*)bump((size_t)NN * EE * 4);
  float*    rowadd  = (float*)bump((size_t)NN * EE * 4);
  float*    dis     = (float*)bump((size_t)NN * 4);
  bf16*     Xb      = (bf16*)bump((size_t)NN * WIN * 2);
  bf16*     WgT     = (bf16*)bump((size_t)WIN * WIN * 2);
  bf16*     Xgb     = (bf16*)bump((size_t)NN * WIN * 2);
  bf16*     weightT = (bf16*)bump((size_t)WIN * WOUT * 2);
  float*    Xwf     = (float*)bump((size_t)NN * WOUT * 4);
  bf16*     XwT     = (bf16*)bump((size_t)NN * WOUT * 2);
  float*    lgcn    = (float*)bump((size_t)EE * NN * WOUT * 4);
  float*    tmpart  = (float*)bump((size_t)EE * 32 * WOUT * 4);
  float*    tm      = (float*)bump((size_t)EE * WOUT * 4);
  float*    beta    = (float*)bump((size_t)EE * 4);
  bf16*     Xcb     = (bf16*)bump((size_t)NN * H0 * 2);
  bf16*     W0T     = (bf16*)bump((size_t)H0 * WOUT * 2);
  float*    x0f     = (float*)bump((size_t)NN * WOUT * 4);
  bf16*     xb      = (bf16*)bump((size_t)NN * WOUT * 2);
  bf16*     xbT     = (bf16*)bump((size_t)NN * WOUT * 2);
  bf16*     WcT     = (bf16*)bump((size_t)NLAYERS * WOUT * WOUT * 2);
  float*    hhf     = (float*)bump((size_t)NN * WOUT * 4);
  bf16*     hhb     = (bf16*)bump((size_t)NN * WOUT * 2);
  bf16*     Wd1T    = (bf16*)bump((size_t)WOUT * WOUT * 2);
  bf16*     W1T     = (bf16*)bump((size_t)WOUT * WOUT * 2);
  bf16*     W2T     = (bf16*)bump((size_t)WOUT * NCLS * 2);
  bf16*     Z1b     = (bf16*)bump((size_t)NN * WOUT * 2);
  bf16*     Z2b     = (bf16*)bump((size_t)NN * WOUT * 2);
  bf16*     Zgb     = (bf16*)bump((size_t)MM * WOUT * 2);
  float*    lrow    = (float*)bump((size_t)MM * 4);

  auto tw = [&](const float* in, bf16* o, int B, int R, int C) {
    size_t tot = (size_t)B * R * C;
    twK<<<dim3((unsigned)((tot + 255) / 256)), 256, 0, stream>>>(in, o, B, R, C);
  };
  tw(Wg, WgT, 1, 512, 512);
  tw(weight, weightT, 1, 512, 128);
  tw(W0, W0T, 1, H0, 128);
  tw(Wconvs, WcT, NLAYERS, 128, 128);
  tw(Wd1, Wd1T, 1, 128, 128);
  tw(W1, W1T, 1, 128, 128);
  tw(W2, W2T, 1, 128, NCLS);

  convK<<<(NN * WIN + 255) / 256, 256, 0, stream>>>(X, Xb, NN * WIN);
  colsumK<<<NN * EE / 256, 256, 0, stream>>>(A, colsum);
  dinvK<<<NN * EE / 256, 256, 0, stream>>>(A, colsum, dinv, rowadd);
  transposeAK<<<dim3(NN / 32, NN / 32), 256, 0, stream>>>(A, At, Bmask);
  degreeK<<<NN, 128, 0, stream>>>(Bmask, dis);
  ahatK<<<NN * 128 / 256, 256, 0, stream>>>(Bmask, dis, Ahat);

  auto gemm = [&](const bf16* Amat, int lda, const bf16* Bt, int ldb,
                  int M, int N, int K, float sacc, float saux, int act,
                  const float* bias_, const float* rs, const float* ra,
                  const float* aux, float* oF, bf16* oB, bf16* oBT) {
    if (N % 64 == 0) {
      int tiles = (M / 16) * (N / 64);
      gemm_bf16_k<4><<<(tiles + 3) / 4, 128, 0, stream>>>(
          Amat, lda, Bt, ldb, M, N, K, sacc, saux, act,
          bias_, rs, ra, aux, oF, oB, oBT);
    } else {
      int tiles = (M / 16) * (N / 16);
      gemm_bf16_k<1><<<(tiles + 3) / 4, 128, 0, stream>>>(
          Amat, lda, Bt, ldb, M, N, K, sacc, saux, act,
          bias_, rs, ra, aux, oF, oB, oBT);
    }
  };

  // Xg = leaky_relu(X@Wg + bg)
  gemm(Xb, WIN, WgT, WIN, NN, WIN, WIN, 1.f, 0.f, 2, bg,
       nullptr, nullptr, nullptr, nullptr, Xgb, nullptr);
  // Xw = leaky_relu(Xg@weight) -> f32 + transposed bf16
  gemm(Xgb, WIN, weightT, WIN, NN, WOUT, WIN, 1.f, 0.f, 2, nullptr,
       nullptr, nullptr, nullptr, Xwf, nullptr, XwT);
  // lgcn[e] = relu(dinv_e * (At_e@Xw + (1-diag_e)*Xw))
  for (int e = 0; e < EE; ++e)
    gemm(At + (size_t)e * NN * NN, NN, XwT, NN, NN, WOUT, NN, 1.f, 0.f, 1, nullptr,
         dinv + e * NN, rowadd + e * NN, Xwf,
         lgcn + (size_t)e * NN * WOUT, nullptr, nullptr);

  tmpartK<<<dim3(EE, 32), 128, 0, stream>>>(lgcn, attw, tmpart);
  tmfinK<<<EE, 128, 0, stream>>>(tmpart, attb, tm);
  betaK<<<1, 128, 0, stream>>>(tm, attq, beta);
  concatK<<<(NN * H0) / 256, 256, 0, stream>>>(lgcn, X, beta, Xcb);

  // x0 = relu(X_@W0 + b0) -> f32 + bf16 + bf16^T (initial x)
  gemm(Xcb, H0, W0T, H0, NN, WOUT, H0, 1.f, 0.f, 1, b0,
       nullptr, nullptr, nullptr, x0f, xb, xbT);

  // GCNII stack: A_hat lives in L2 across all 64 layers
  for (int l = 0; l < NLAYERS; ++l) {
    float bl = logf(0.5f / (float)(l + 1) + 1.0f);
    // hh = 0.9*(A_hat@x) + 0.1*x0
    gemm(Ahat, NN, xbT, NN, NN, WOUT, NN, 0.9f, 0.1f, 0, nullptr,
         nullptr, nullptr, x0f, hhf, hhb, nullptr);
    // x = relu((1-bl)*hh + bl*(hh@Wl)) -> bf16 + bf16^T
    gemm(hhb, WOUT, WcT + (size_t)l * WOUT * WOUT, WOUT, NN, WOUT, WOUT,
         bl, 1.f - bl, 1, nullptr, nullptr, nullptr, hhf, nullptr, xb, xbT);
  }

  // Z = leaky_relu(x@Wd1 + bd1); Z = leaky_relu(Z@W1 + b1)
  gemm(xb, WOUT, Wd1T, WOUT, NN, WOUT, WOUT, 1.f, 0.f, 2, bd1,
       nullptr, nullptr, nullptr, nullptr, Z1b, nullptr);
  gemm(Z1b, WOUT, W1T, WOUT, NN, WOUT, WOUT, 1.f, 0.f, 2, b1,
       nullptr, nullptr, nullptr, nullptr, Z2b, nullptr);

  // y = Z[target_x]@W2 + b2  -> out[1..]
  gatherK<<<(MM * WOUT) / 256, 256, 0, stream>>>(Z2b, tgtx, Zgb);
  gemm(Zgb, WOUT, W2T, WOUT, MM, NCLS, WOUT, 1.f, 0.f, 0, b2,
       nullptr, nullptr, nullptr, out + 1, nullptr, nullptr);

  // loss = -mean(log_softmax(y)[target])  -> out[0]
  lossrowK<<<MM / 256, 256, 0, stream>>>(out + 1, tgt, lrow);
  loss2K<<<1, 256, 0, stream>>>(lrow, out);
}
